// HeteAttentionHead_48284022342203
// MI455X (gfx1250) — matrix operand order repaired and checked by reading, not compiled, not run on hardware
//
#include <hip/hip_runtime.h>
#include <hip/hip_bf16.h>

#define N_SRC 16384
#define N_NBR 16384
#define E_EDGES 524288
#define DIM 128
#define ALPHA 0.2f

typedef __attribute__((ext_vector_type(16))) _Float16 v16h;
typedef __attribute__((ext_vector_type(8)))  _Float16 v8h;
typedef __attribute__((ext_vector_type(8)))  float    v8f;

// LDS row stride in halves (pad 128 -> 136 to break bank conflicts; keeps 16B align)
#define WSTR 136
// Rows per block in the GEMM (4 M-tiles of 16)
#define MROWS 64

// ---------------------------------------------------------------------------
// Zero-init: out (N_SRC*DIM) and seg_sum (N_SRC). Required every launch since
// we accumulate with atomics and the harness does not re-poison buffers.
// ---------------------------------------------------------------------------
__global__ __launch_bounds__(256) void init_zero_kernel(float* __restrict__ out,
                                                        float* __restrict__ seg) {
    const int total = N_SRC * DIM;
    for (int i = blockIdx.x * 256 + threadIdx.x; i < total; i += gridDim.x * 256)
        out[i] = 0.0f;
    for (int i = blockIdx.x * 256 + threadIdx.x; i < N_SRC; i += gridDim.x * 256)
        seg[i] = 0.0f;
}

// ---------------------------------------------------------------------------
// Y[M x 128] = X[M x 128] @ W[128 x 128] + b      (M = 16384)
// Block: 256 threads = 8 wave32s; block computes 64 rows x 128 cols.
// Wave w owns columns [16w, 16w+16) and 4 M-tiles (4 f32 accumulators).
// Per K-step the B fragment is loaded once from LDS and reused across the
// 4 WMMAs -> 16 v_wmma_f32_16x16x32_f16 per wave, K=128 in 4 steps.
// W is staged transposed ([n][k]) in LDS as f16 so B fragments are contiguous.
// ---------------------------------------------------------------------------
__global__ __launch_bounds__(256) void mlp_gemm_kernel(const float* __restrict__ X,
                                                       const float* __restrict__ W,
                                                       const float* __restrict__ b,
                                                       float* __restrict__ Y) {
    __shared__ _Float16 Wl[DIM * WSTR];     // transposed: Wl[n*WSTR + k] = W[k][n]
    __shared__ _Float16 Xl[MROWS * WSTR];   // 64-row A tile, f16

    const int tid  = threadIdx.x;
    const int row0 = blockIdx.x * MROWS;

    // Stage + convert W (transposed) and the X tile.
    for (int i = tid; i < DIM * DIM; i += 256) {
        int k = i >> 7, n = i & 127;
        Wl[n * WSTR + k] = (_Float16)W[i];
    }
    for (int i = tid; i < MROWS * DIM; i += 256) {
        int r = i >> 7, c = i & 127;
        Xl[r * WSTR + c] = (_Float16)X[(size_t)(row0 + r) * DIM + c];
    }
    __syncthreads();

    const int wave = tid >> 5;        // 0..7 -> N tile
    const int lane = tid & 31;
    const int half = lane >> 4;       // 0: lanes 0-15, 1: lanes 16-31
    const int idx  = lane & 15;       // row (A) / col (B,C) within tile
    const int col0 = wave * 16;

    v8f acc0 = {}, acc1 = {}, acc2 = {}, acc3 = {};
    #pragma unroll
    for (int kt = 0; kt < 4; ++kt) {
        // B fragment (32x16 f16): lanes 0-15 hold K=0..15, lanes 16-31 hold K=16..31
        const int kb = kt * 32 + half * 16;
        v8h blo = *(const v8h*)&Wl[(col0 + idx) * WSTR + kb];
        v8h bhi = *(const v8h*)&Wl[(col0 + idx) * WSTR + kb + 8];
        v16h bm = __builtin_shufflevector(blo, bhi, 0,1,2,3,4,5,6,7,8,9,10,11,12,13,14,15);

        // A fragments (16x32 f16): lane half selects K 0..15 / 16..31
        const int ka = kt * 32 + half * 16;
        #pragma unroll
        for (int mt = 0; mt < 4; ++mt) {
            const int ar = (mt * 16 + idx) * WSTR + ka;
            v8h alo = *(const v8h*)&Xl[ar];
            v8h ahi = *(const v8h*)&Xl[ar + 8];
            v16h a = __builtin_shufflevector(alo, ahi, 0,1,2,3,4,5,6,7,8,9,10,11,12,13,14,15);
            v8f& acc = (mt == 0) ? acc0 : (mt == 1) ? acc1 : (mt == 2) ? acc2 : acc3;
            acc = __builtin_amdgcn_wmma_f32_16x16x32_f16(
                /*neg_a=*/false, a, /*neg_b=*/false, bm,
                /*c_mod=*/(short)0, acc, /*reuse_a=*/false, /*reuse_b=*/false);
        }
    }

    const float bias = b[col0 + idx];
    // C/D layout: VGPR r -> M = r + 8*half, N = col0 + idx
    #pragma unroll
    for (int mt = 0; mt < 4; ++mt) {
        const v8f& acc = (mt == 0) ? acc0 : (mt == 1) ? acc1 : (mt == 2) ? acc2 : acc3;
        #pragma unroll
        for (int r = 0; r < 8; ++r) {
            int m = mt * 16 + r + half * 8;
            Y[(size_t)(row0 + m) * DIM + col0 + idx] = acc[r] + bias;
        }
    }
}

// ---------------------------------------------------------------------------
// Per-node attention scalars: a_cur[i] = dot(cur[i], Wa[0:128]),
//                             a_nbr[j] = dot(nbr[j], Wa[128:256]).
// One wave32 per row; lane handles a float4; shuffle reduction.
// ---------------------------------------------------------------------------
__global__ __launch_bounds__(256) void node_dots_kernel(const float* __restrict__ cur,
                                                        const float* __restrict__ nbr,
                                                        const float* __restrict__ Wa,
                                                        float* __restrict__ a_cur,
                                                        float* __restrict__ a_nbr) {
    const int wave = threadIdx.x >> 5;
    const int lane = threadIdx.x & 31;
    const int row  = blockIdx.x * 8 + wave;        // 0 .. 2*N-1
    if (row >= 2 * N_SRC) return;

    const bool is_cur = (row < N_SRC);
    const int  r      = is_cur ? row : row - N_SRC;
    const float* rowp = is_cur ? (cur + (size_t)r * DIM) : (nbr + (size_t)r * DIM);
    const float* wap  = is_cur ? Wa : (Wa + DIM);

    float4 v = ((const float4*)rowp)[lane];
    float4 w = ((const float4*)wap)[lane];
    float s = v.x * w.x + v.y * w.y + v.z * w.z + v.w * w.w;
    #pragma unroll
    for (int off = 16; off > 0; off >>= 1) s += __shfl_down(s, off, 32);
    if (lane == 0) {
        if (is_cur) a_cur[r] = s; else a_nbr[r] = s;
    }
}

// ---------------------------------------------------------------------------
// Edge pass 1: s = leaky_relu(a_cur[src] + a_nbr[dst] + ba); ex = exp(s);
// expbuf[e] = ex; seg[src] += ex (hardware f32 atomic, lands in L2).
// ---------------------------------------------------------------------------
__global__ __launch_bounds__(256) void edge_scores_kernel(const int2* __restrict__ edges,
                                                          const float* __restrict__ a_cur,
                                                          const float* __restrict__ a_nbr,
                                                          const float* __restrict__ ba,
                                                          float* __restrict__ expbuf,
                                                          float* __restrict__ seg) {
    const int e = blockIdx.x * 256 + threadIdx.x;
    if (e >= E_EDGES) return;
    const int2 ed = edges[e];            // global_load_b64
    float s = a_cur[ed.x] + a_nbr[ed.y] + ba[0];
    s = (s > 0.0f) ? s : ALPHA * s;
    const float ex = __expf(s);
    expbuf[e] = ex;
    unsafeAtomicAdd(&seg[ed.x], ex);
}

// ---------------------------------------------------------------------------
// Edge pass 2 (dominant phase): out[src] += (expbuf[e]/seg[src]) * nbr[dst].
// One wave32 per edge; lane owns a float4 slice. nbr (8MB) and out (8MB) both
// resident in the 192MB L2, so gathers + f32 atomics run at L2 bandwidth.
// ---------------------------------------------------------------------------
__global__ __launch_bounds__(256) void aggregate_kernel(const int2* __restrict__ edges,
                                                        const float* __restrict__ expbuf,
                                                        const float* __restrict__ seg,
                                                        const float* __restrict__ nbr,
                                                        float* __restrict__ out) {
    const int wave = threadIdx.x >> 5;
    const int lane = threadIdx.x & 31;
    const int e = blockIdx.x * 8 + wave;
    if (e >= E_EDGES) return;

    const int2 ed = edges[e];
    const float w = expbuf[e] / seg[ed.x];

    float4 v = ((const float4*)(nbr + (size_t)ed.y * DIM))[lane];  // global_load_b128
    float* o = out + (size_t)ed.x * DIM + lane * 4;
    unsafeAtomicAdd(o + 0, w * v.x);
    unsafeAtomicAdd(o + 1, w * v.y);
    unsafeAtomicAdd(o + 2, w * v.z);
    unsafeAtomicAdd(o + 3, w * v.w);
}

// ---------------------------------------------------------------------------
extern "C" void kernel_launch(void* const* d_in, const int* in_sizes, int n_in,
                              void* d_out, int out_size, void* d_ws, size_t ws_size,
                              hipStream_t stream) {
    const float* x_cur = (const float*)d_in[0];
    const float* x_nbr = (const float*)d_in[1];
    const float* W1c   = (const float*)d_in[2];
    const float* b1c   = (const float*)d_in[3];
    const float* W2c   = (const float*)d_in[4];
    const float* b2c   = (const float*)d_in[5];
    const float* W1n   = (const float*)d_in[6];
    const float* b1n   = (const float*)d_in[7];
    const float* W2n   = (const float*)d_in[8];
    const float* b2n   = (const float*)d_in[9];
    const float* Wa    = (const float*)d_in[10];
    const float* ba    = (const float*)d_in[11];
    const int2*  edges = (const int2*)d_in[12];
    float* out = (float*)d_out;

    // Workspace layout (~27.5 MB)
    char* p = (char*)d_ws;
    float* h      = (float*)p; p += (size_t)N_SRC * DIM * sizeof(float);   // 8 MB (reused)
    float* cur    = (float*)p; p += (size_t)N_SRC * DIM * sizeof(float);   // 8 MB
    float* nbr    = (float*)p; p += (size_t)N_NBR * DIM * sizeof(float);   // 8 MB
    float* a_cur  = (float*)p; p += (size_t)N_SRC * sizeof(float);
    float* a_nbr  = (float*)p; p += (size_t)N_NBR * sizeof(float);
    float* seg    = (float*)p; p += (size_t)N_SRC * sizeof(float);
    float* expbuf = (float*)p; p += (size_t)E_EDGES * sizeof(float);       // 2 MB

    init_zero_kernel<<<2048, 256, 0, stream>>>(out, seg);

    // Two-layer MLPs via WMMA GEMMs (h reused; same-stream ordering serializes)
    mlp_gemm_kernel<<<N_SRC / MROWS, 256, 0, stream>>>(x_cur, W1c, b1c, h);
    mlp_gemm_kernel<<<N_SRC / MROWS, 256, 0, stream>>>(h,     W2c, b2c, cur);
    mlp_gemm_kernel<<<N_NBR / MROWS, 256, 0, stream>>>(x_nbr, W1n, b1n, h);
    mlp_gemm_kernel<<<N_NBR / MROWS, 256, 0, stream>>>(h,     W2n, b2n, nbr);

    node_dots_kernel<<<(2 * N_SRC) / 8, 256, 0, stream>>>(cur, nbr, Wa, a_cur, a_nbr);
    edge_scores_kernel<<<E_EDGES / 256, 256, 0, stream>>>(edges, a_cur, a_nbr, ba, expbuf, seg);
    aggregate_kernel<<<E_EDGES / 8, 256, 0, stream>>>(edges, expbuf, seg, nbr, out);
}
